// DeepseekV2DecoderLayer_9672266350854
// MI455X (gfx1250) — matrix operand order repaired and checked by reading, not compiled, not run on hardware
//
#include <hip/hip_runtime.h>
#include <cstdint>
#include <cstddef>

// ---------------------------------------------------------------------------
// DeepSeek-V2 decoder layer for MI455X (gfx1250), wave32 + WMMA bf16.
// Heavy GEMMs use v_wmma_f32_16x16x32_bf16 with the B-tile staged in LDS via
// GLOBAL_LOAD_ASYNC_TO_LDS_B128 (ASYNCcnt-tracked, double-buffered) so the
// next K-slab streams into LDS while the current one feeds the WMMA pipe.
// Weights are converted to bf16 once per call (~107MB -> resident in 192MB L2).
// ---------------------------------------------------------------------------

#define H_DIM   2048
#define NH      16
#define D_NOPE  64
#define D_ROPE  64
#define D_V     64
#define QHEAD   128          // D_NOPE + D_ROPE
#define KV_RANK 256
#define NE      8
#define INTER   1408
#define SEQ     2048

typedef __attribute__((ext_vector_type(16))) __bf16 v16bf;
typedef __attribute__((ext_vector_type(8)))  __bf16 bf16x8;
typedef __attribute__((ext_vector_type(8)))  float  v8f;

union FragBF { v16bf v; bf16x8 h[2]; };

// Async global->LDS copy, 16B per lane. LDS address = low 32 bits of the flat
// shared pointer (ISA: LDS aperture truncates addr[31:0]).
__device__ __forceinline__ void async_copy_b128(const void* lds_dst, const void* gsrc) {
    asm volatile("global_load_async_to_lds_b128 %0, %1, off"
                 :: "v"((uint32_t)(uintptr_t)lds_dst),
                    "v"((uint64_t)(uintptr_t)gsrc)
                 : "memory");
}
__device__ __forceinline__ void wait_async_le1() {
    asm volatile("s_wait_asynccnt 0x1" ::: "memory");
}
__device__ __forceinline__ void wait_async_le0() {
    asm volatile("s_wait_asynccnt 0x0" ::: "memory");
}

// -------------------------------- helpers ----------------------------------

__global__ void cvt_f32_bf16(const float* __restrict__ in, __bf16* __restrict__ out, int n) {
    int i = blockIdx.x * blockDim.x + threadIdx.x;
    if (i < n) out[i] = (__bf16)in[i];
}

// RMSNorm over `len` elements of each row; bf16 output.
__global__ __launch_bounds__(256)
void rmsnorm_bf16(const float* __restrict__ x, const float* __restrict__ w,
                  __bf16* __restrict__ out, int ldin, int len, int ldout) {
    const int row = blockIdx.x;
    const float* xr = x + (size_t)row * ldin;
    __shared__ float red[256];
    float s = 0.f;
    for (int i = threadIdx.x; i < len; i += 256) { float v = xr[i]; s += v * v; }
    red[threadIdx.x] = s;
    __syncthreads();
    for (int o = 128; o > 0; o >>= 1) {
        if (threadIdx.x < o) red[threadIdx.x] += red[threadIdx.x + o];
        __syncthreads();
    }
    const float inv = rsqrtf(red[0] / (float)len + 1e-6f);
    for (int i = threadIdx.x; i < len; i += 256)
        out[(size_t)row * ldout + i] = (__bf16)(w[i] * xr[i] * inv);
}

// ------------------------------ WMMA GEMM ----------------------------------
// C[M][N] (f32, ldc) = A[M][K] (bf16, lda, row-major) * B[N][K]^T (bf16, ldb)
// 256 threads = 8 waves. Macro tile 128(M) x 64(N); wave w owns rows
// [by*128 + w*16, +16), 4 accumulators across 64 N-columns. B tile is
// double-buffered in LDS, filled by async global->LDS B128 copies so the
// next K-slab is in flight while WMMAs consume the current one. Rows padded
// to 48 halves (96B): 16B-aligned ds_load_b128, low bank conflict.
template <bool ACC>
__global__ __launch_bounds__(256)
void gemm_bf16(const __bf16* __restrict__ A, const __bf16* __restrict__ B,
               float* __restrict__ C, int M, int N, int K,
               int lda, int ldb, int ldc) {
    __shared__ __bf16 bt[2][64][48];
    const int tid = threadIdx.x, wave = tid >> 5, lane = tid & 31;
    const int l15 = lane & 15, hi = lane >> 4;
    const int m0 = blockIdx.y * 128 + wave * 16;
    const int n0 = blockIdx.x * 64;

    const int ldr = tid >> 2;            // B-tile row this thread fills
    const int ldc8 = (tid & 3) * 8;      // ... and its 16B chunk
    const __bf16* bsrc = B + (size_t)(n0 + ldr) * ldb + ldc8;

    auto issue = [&](int kt, int buf) {
        async_copy_b128(&bt[buf][ldr][ldc8], bsrc + kt * 32);
    };

    v8f acc[4] = {};
    const int ntiles = K / 32;
    issue(0, 0);

    for (int kt = 0; kt < ntiles; ++kt) {
        const int cur = kt & 1;
        if (kt + 1 < ntiles) { issue(kt + 1, cur ^ 1); wait_async_le1(); }
        else                 { wait_async_le0(); }
        __syncthreads();   // all waves' async fills of bt[cur] complete

        // A fragment: 16x32 bf16 per ISA layout (lane<16: K {0..7,16..23}).
        FragBF a;
        const __bf16* arow = A + (size_t)(m0 + l15) * lda + kt * 32;
        a.h[0] = *(const bf16x8*)(arow + hi * 8);
        a.h[1] = *(const bf16x8*)(arow + 16 + hi * 8);

#pragma unroll
        for (int t = 0; t < 4; ++t) {
            FragBF b;  // B fragment: lane<16 -> N=l15, K {0..7,16..23}.
            const __bf16* brow = &bt[cur][t * 16 + l15][hi * 8];
            b.h[0] = *(const bf16x8*)(brow);
            b.h[1] = *(const bf16x8*)(brow + 16);
            acc[t] = __builtin_amdgcn_wmma_f32_16x16x32_bf16(
                false, a.v, false, b.v, (short)0, acc[t], false, false);
        }
        __syncthreads();   // bt[cur] free for reuse two iterations later
    }

    // C layout: vgpr r -> row r + 8*(lane>=16), col = lane&15.
#pragma unroll
    for (int t = 0; t < 4; ++t) {
#pragma unroll
        for (int r = 0; r < 8; ++r) {
            const int row = m0 + r + 8 * hi;
            const int col = n0 + t * 16 + l15;
            const size_t idx = (size_t)row * ldc + col;
            float v = acc[t][r];
            if (ACC) v += C[idx];
            C[idx] = v;
        }
    }
}

// --------------------------- RoPE / QKV prepare ----------------------------
// q_f32 [S][NH*128] -> qb [NH][S][128] (rope on last 64)
// kv_f32[S][NH*128] -> kb [NH][S][128] (nope||roped k_pe), vt [NH][64][S]
__global__ __launch_bounds__(128)
void prepare_qkv(const float* __restrict__ qf, const float* __restrict__ kvf,
                 const float* __restrict__ ckv,
                 __bf16* __restrict__ qb, __bf16* __restrict__ kb,
                 __bf16* __restrict__ vt) {
    const int s = blockIdx.x, h = blockIdx.y, d = threadIdx.x;  // d in [0,128)
    const size_t qidx = (size_t)s * (NH * QHEAD) + h * QHEAD;
    float outq, outk;
    if (d < D_NOPE) {
        outq = qf[qidx + d];
        outk = kvf[qidx + d];            // k_nope
    } else {
        const int dr = d - D_NOPE;       // rope dim 0..63
        const float inv_freq = __powf(10000.0f, -(float)(2 * (dr & 31)) / (float)D_ROPE);
        const float ang = (float)s * inv_freq;
        const float c = __cosf(ang), sn = __sinf(ang);
        const float x = qf[qidx + D_NOPE + dr];
        const float xr = (dr < 32) ? -qf[qidx + D_NOPE + dr + 32]
                                   :  qf[qidx + D_NOPE + dr - 32];
        outq = x * c + xr * sn;
        const float kp  = ckv[(size_t)s * (KV_RANK + D_ROPE) + KV_RANK + dr];
        const float kpr = (dr < 32) ? -ckv[(size_t)s * (KV_RANK + D_ROPE) + KV_RANK + dr + 32]
                                    :  ckv[(size_t)s * (KV_RANK + D_ROPE) + KV_RANK + dr - 32];
        outk = kp * c + kpr * sn;        // broadcast k_pe per head
    }
    qb[((size_t)h * SEQ + s) * QHEAD + d] = (__bf16)outq;
    kb[((size_t)h * SEQ + s) * QHEAD + d] = (__bf16)outk;
    if (d < D_V)  // v transposed: [h][d][s] so P*V B-fragments load contiguously
        vt[((size_t)h * D_V + d) * SEQ + s] = (__bf16)kvf[qidx + D_NOPE + d];
}

// ------------------------- flash attention (causal) ------------------------
// grid (S/128, NH), 8 waves; wave owns 16 query rows, K tiled by 32.
__global__ __launch_bounds__(256)
void attention(const __bf16* __restrict__ Qh, const __bf16* __restrict__ Kh,
               const __bf16* __restrict__ Vt, __bf16* __restrict__ ctx) {
    const int h = blockIdx.y;
    const int tid = threadIdx.x, wave = tid >> 5, lane = tid & 31;
    const int l15 = lane & 15, hi = lane >> 4;
    const int q0 = blockIdx.x * 128 + wave * 16;

    const __bf16* Q  = Qh + (size_t)h * SEQ * QHEAD;
    const __bf16* Kp = Kh + (size_t)h * SEQ * QHEAD;
    const __bf16* Vp = Vt + (size_t)h * D_V * SEQ;

    __shared__ __bf16 plds[8][16][48];  // per-wave P tile (16x32, padded rows)

    // Q fragments over d = 0..127 (4 chunks of 32)
    FragBF qf[4];
    {
        const __bf16* qrow = Q + (size_t)(q0 + l15) * QHEAD;
#pragma unroll
        for (int d = 0; d < 4; ++d) {
            qf[d].h[0] = *(const bf16x8*)(qrow + d * 32 + hi * 8);
            qf[d].h[1] = *(const bf16x8*)(qrow + d * 32 + 16 + hi * 8);
        }
    }

    v8f acc[4] = {};           // ctx accumulators over v = 0..63
    float m[8], l[8];
#pragma unroll
    for (int r = 0; r < 8; ++r) { m[r] = -3.0e38f; l[r] = 0.f; }
    const float scale = 0.08838834764831845f;  // 128^-0.5

    const int kend = q0 + 16;  // last key tile touching the diagonal
    for (int kt = 0; kt < kend; kt += 32) {
        // ---- scores: two 16x16 C fragments covering keys kt..kt+31 ----
        v8f s0 = {}, s1 = {};
#pragma unroll
        for (int d = 0; d < 4; ++d) {
            FragBF b0, b1;  // B = K^T: lane<16 -> key kt+l15, d {0..7,16..23}
            const __bf16* k0row = Kp + (size_t)(kt + l15) * QHEAD + d * 32 + hi * 8;
            b0.h[0] = *(const bf16x8*)(k0row);
            b0.h[1] = *(const bf16x8*)(k0row + 16);
            const __bf16* k1row = Kp + (size_t)(kt + 16 + l15) * QHEAD + d * 32 + hi * 8;
            b1.h[0] = *(const bf16x8*)(k1row);
            b1.h[1] = *(const bf16x8*)(k1row + 16);
            s0 = __builtin_amdgcn_wmma_f32_16x16x32_bf16(false, qf[d].v, false, b0.v,
                                                         (short)0, s0, false, false);
            s1 = __builtin_amdgcn_wmma_f32_16x16x32_bf16(false, qf[d].v, false, b1.v,
                                                         (short)0, s1, false, false);
        }

        // ---- online softmax (row reductions within 16-lane halves) ----
#pragma unroll
        for (int r = 0; r < 8; ++r) {
            const int q = q0 + r + 8 * hi;
            float a0 = s0[r] * scale, a1 = s1[r] * scale;
            if (kt + l15 > q)       a0 = -3.0e38f;
            if (kt + 16 + l15 > q)  a1 = -3.0e38f;
            float mx = fmaxf(a0, a1);
#pragma unroll
            for (int o = 8; o; o >>= 1) mx = fmaxf(mx, __shfl_xor(mx, o, 32));
            const float mnew = fmaxf(m[r], mx);
            const float corr = __expf(m[r] - mnew);
            m[r] = mnew;
            l[r] *= corr;
#pragma unroll
            for (int t = 0; t < 4; ++t) acc[t][r] *= corr;
            const float p0 = __expf(a0 - mnew);
            const float p1 = __expf(a1 - mnew);
            l[r] += p0 + p1;  // per-lane partial row sum
            plds[wave][r + 8 * hi][l15]      = (__bf16)p0;
            plds[wave][r + 8 * hi][l15 + 16] = (__bf16)p1;
        }
        asm volatile("s_wait_dscnt 0" ::: "memory");  // LDS C->A layout fixup

        // ---- P (16x32) * V (32x64): 4 bf16 WMMAs ----
        FragBF pf;
        const __bf16* prow = &plds[wave][l15][0];
        pf.h[0] = *(const bf16x8*)(prow + hi * 8);
        pf.h[1] = *(const bf16x8*)(prow + 16 + hi * 8);
#pragma unroll
        for (int t = 0; t < 4; ++t) {
            FragBF vb;  // V^T rows in Vt: lane<16 -> v-col l15, keys {0..7,16..23}
            const __bf16* vrow = Vp + (size_t)(t * 16 + l15) * SEQ + kt + hi * 8;
            vb.h[0] = *(const bf16x8*)(vrow);
            vb.h[1] = *(const bf16x8*)(vrow + 16);
            acc[t] = __builtin_amdgcn_wmma_f32_16x16x32_bf16(
                false, pf.v, false, vb.v, (short)0, acc[t], false, false);
        }
    }

    // ---- finalize: divide by row sum, write ctx [S][NH*64] ----
#pragma unroll
    for (int r = 0; r < 8; ++r) {
        float ls = l[r];
#pragma unroll
        for (int o = 8; o; o >>= 1) ls += __shfl_xor(ls, o, 32);
        const float inv = 1.0f / ls;
        const int q = q0 + r + 8 * hi;
#pragma unroll
        for (int t = 0; t < 4; ++t)
            ctx[(size_t)q * (NH * D_V) + h * D_V + t * 16 + l15] =
                (__bf16)(acc[t][r] * inv);
    }
}

// ------------------------------ MoE pieces ---------------------------------

__global__ __launch_bounds__(256)
void gate_softmax(const __bf16* __restrict__ mlp_in, const float* __restrict__ gw,
                  float* __restrict__ probs) {
    const int s = blockIdx.x, wave = threadIdx.x >> 5, lane = threadIdx.x & 31;
    const __bf16* xr = mlp_in + (size_t)s * H_DIM;
    const float* w = gw + (size_t)wave * H_DIM;
    float acc = 0.f;
    for (int i = lane; i < H_DIM; i += 32) acc += (float)xr[i] * w[i];
#pragma unroll
    for (int o = 16; o; o >>= 1) acc += __shfl_xor(acc, o, 32);
    __shared__ float lg[NE];
    if (lane == 0) lg[wave] = acc;
    __syncthreads();
    if (threadIdx.x < NE) {
        float mx = -3.0e38f;
        for (int e = 0; e < NE; ++e) mx = fmaxf(mx, lg[e]);
        float sum = 0.f;
        for (int e = 0; e < NE; ++e) sum += __expf(lg[e] - mx);
        probs[(size_t)s * NE + threadIdx.x] = __expf(lg[threadIdx.x] - mx) / sum;
    }
}

__global__ void silu_scale(const float* __restrict__ fc1o, const float* __restrict__ probs,
                           __bf16* __restrict__ hb, int e) {
    const size_t i = (size_t)blockIdx.x * blockDim.x + threadIdx.x;
    if (i >= (size_t)SEQ * INTER) return;
    const int s = (int)(i / INTER);
    const float x = fc1o[i];
    hb[i] = (__bf16)(probs[(size_t)s * NE + e] * (x / (1.0f + __expf(-x))));
}

// --------------------------------- launch ----------------------------------

extern "C" void kernel_launch(void* const* d_in, const int* in_sizes, int n_in,
                              void* d_out, int out_size, void* d_ws, size_t ws_size,
                              hipStream_t stream) {
    (void)in_sizes; (void)n_in; (void)out_size; (void)ws_size;
    const float* hidden = (const float*)d_in[1];
    const float* ln1    = (const float*)d_in[2];
    const float* wq_f   = (const float*)d_in[3];
    const float* wkva_f = (const float*)d_in[4];
    const float* lnkv   = (const float*)d_in[5];
    const float* wkvb_f = (const float*)d_in[6];
    const float* wo_f   = (const float*)d_in[7];
    const float* ln2    = (const float*)d_in[8];
    const float* wg     = (const float*)d_in[9];
    const float* wf1_f  = (const float*)d_in[10];
    const float* wf2_f  = (const float*)d_in[11];
    float* out = (float*)d_out;

    // bump allocator over d_ws (total ~187 MB incl. aliasing)
    char* p = (char*)d_ws;
    auto alloc = [&](size_t bytes) -> char* {
        char* r = p;
        p += (bytes + 255) & ~(size_t)255;
        return r;
    };
    __bf16* wq   = (__bf16*)alloc((size_t)NH * QHEAD * H_DIM * 2);
    __bf16* wkva = (__bf16*)alloc((size_t)(KV_RANK + D_ROPE) * H_DIM * 2);
    __bf16* wkvb = (__bf16*)alloc((size_t)NH * (D_NOPE + D_V) * KV_RANK * 2);
    __bf16* wo   = (__bf16*)alloc((size_t)H_DIM * NH * D_V * 2);
    __bf16* wf1  = (__bf16*)alloc((size_t)NE * INTER * H_DIM * 2);
    __bf16* wf2  = (__bf16*)alloc((size_t)NE * H_DIM * INTER * 2);
    __bf16* sa_in = (__bf16*)alloc((size_t)SEQ * H_DIM * 2);   // aliased as h_bf later
    float*  q_f32 = (float*)alloc((size_t)SEQ * NH * QHEAD * 4); // aliased as x1 later
    __bf16* q_bf  = (__bf16*)alloc((size_t)NH * SEQ * QHEAD * 2);
    float*  ckv   = (float*)alloc((size_t)SEQ * (KV_RANK + D_ROPE) * 4);
    __bf16* ckvln = (__bf16*)alloc((size_t)SEQ * KV_RANK * 2);
    float*  kv_f32 = (float*)alloc((size_t)SEQ * NH * (D_NOPE + D_V) * 4); // alias fc1_out
    __bf16* k_bf  = (__bf16*)alloc((size_t)NH * SEQ * QHEAD * 2);
    __bf16* v_t   = (__bf16*)alloc((size_t)NH * D_V * SEQ * 2);
    __bf16* ctx   = (__bf16*)alloc((size_t)SEQ * NH * D_V * 2);
    __bf16* mlp_in = (__bf16*)alloc((size_t)SEQ * H_DIM * 2);
    float*  probs  = (float*)alloc((size_t)SEQ * NE * 4);

    float*  x1      = q_f32;            // alias: q_f32 dead after prepare_qkv
    float*  fc1_out = kv_f32;           // alias: kv_f32 dead after prepare_qkv
    __bf16* h_bf    = sa_in;            // alias: sa_in dead after kv_a/q GEMMs

    auto cvt = [&](const float* src, __bf16* dst, size_t n) {
        cvt_f32_bf16<<<(unsigned)((n + 255) / 256), 256, 0, stream>>>(src, dst, (int)n);
    };
    // weight pre-conversion (L2-resident bf16 copies)
    cvt(wq_f,   wq,   (size_t)NH * QHEAD * H_DIM);
    cvt(wkva_f, wkva, (size_t)(KV_RANK + D_ROPE) * H_DIM);
    cvt(wkvb_f, wkvb, (size_t)NH * (D_NOPE + D_V) * KV_RANK);
    cvt(wo_f,   wo,   (size_t)H_DIM * NH * D_V);
    cvt(wf1_f,  wf1,  (size_t)NE * INTER * H_DIM);
    cvt(wf2_f,  wf2,  (size_t)NE * H_DIM * INTER);

    // 1) input RMSNorm
    rmsnorm_bf16<<<SEQ, 256, 0, stream>>>(hidden, ln1, sa_in, H_DIM, H_DIM, H_DIM);

    // 2) q / kv_a projections
    gemm_bf16<false><<<dim3((NH * QHEAD) / 64, SEQ / 128), 256, 0, stream>>>(
        sa_in, wq, q_f32, SEQ, NH * QHEAD, H_DIM, H_DIM, H_DIM, NH * QHEAD);
    gemm_bf16<false><<<dim3((KV_RANK + D_ROPE) / 64, SEQ / 128), 256, 0, stream>>>(
        sa_in, wkva, ckv, SEQ, KV_RANK + D_ROPE, H_DIM, H_DIM, H_DIM, KV_RANK + D_ROPE);

    // 3) kv_a RMSNorm + kv_b projection
    rmsnorm_bf16<<<SEQ, 256, 0, stream>>>(ckv, lnkv, ckvln, KV_RANK + D_ROPE, KV_RANK, KV_RANK);
    gemm_bf16<false><<<dim3((NH * (D_NOPE + D_V)) / 64, SEQ / 128), 256, 0, stream>>>(
        ckvln, wkvb, kv_f32, SEQ, NH * (D_NOPE + D_V), KV_RANK, KV_RANK, KV_RANK,
        NH * (D_NOPE + D_V));

    // 4) RoPE + head-major layouts (+ transposed V)
    prepare_qkv<<<dim3(SEQ, NH), 128, 0, stream>>>(q_f32, kv_f32, ckv, q_bf, k_bf, v_t);

    // 5) causal flash attention
    attention<<<dim3(SEQ / 128, NH), 256, 0, stream>>>(q_bf, k_bf, v_t, ctx);

    // 6) o_proj with residual (x1 = hidden; x1 += ctx @ Wo^T)
    hipMemcpyAsync(x1, hidden, (size_t)SEQ * H_DIM * 4, hipMemcpyDeviceToDevice, stream);
    gemm_bf16<true><<<dim3(H_DIM / 64, SEQ / 128), 256, 0, stream>>>(
        ctx, wo, x1, SEQ, H_DIM, NH * D_V, NH * D_V, NH * D_V, H_DIM);

    // 7) post RMSNorm + gate softmax
    rmsnorm_bf16<<<SEQ, 256, 0, stream>>>(x1, ln2, mlp_in, H_DIM, H_DIM, H_DIM);
    gate_softmax<<<SEQ, 256, 0, stream>>>(mlp_in, wg, probs);

    // 8) MoE: out = x1 + sum_e probs_e * (silu(mlp_in @ fc1_e^T) @ fc2_e^T)
    hipMemcpyAsync(out, x1, (size_t)SEQ * H_DIM * 4, hipMemcpyDeviceToDevice, stream);
    for (int e = 0; e < NE; ++e) {
        gemm_bf16<false><<<dim3(INTER / 64, SEQ / 128), 256, 0, stream>>>(
            mlp_in, wf1 + (size_t)e * INTER * H_DIM, fc1_out,
            SEQ, INTER, H_DIM, H_DIM, H_DIM, INTER);
        silu_scale<<<(unsigned)(((size_t)SEQ * INTER + 255) / 256), 256, 0, stream>>>(
            fc1_out, probs, h_bf, e);
        gemm_bf16<true><<<dim3(H_DIM / 64, SEQ / 128), 256, 0, stream>>>(
            h_bf, wf2 + (size_t)e * H_DIM * INTER, out,
            SEQ, H_DIM, INTER, INTER, INTER, H_DIM);
    }
}
